// GatedDeltaRuleModule_87608742903896
// MI455X (gfx1250) — compile-verified, matrix operand order checked by reading, not run on hardware
//
#include <hip/hip_runtime.h>

#define B_ 4
#define S_ 4096
#define H_ 16
#define D_ 128
#define T_ 32
#define NC_ (S_ / T_)
#define MSPLIT 4
#define MCOLS (D_ / MSPLIT)     // 32 v-columns per block
#define NWAVES (MCOLS / 16)     // 2 waves per block
#define NTHREADS (NWAVES * 32)  // 64 threads

typedef __attribute__((ext_vector_type(16))) __bf16 v16bf;
typedef __attribute__((ext_vector_type(8)))  float  v8f;

union HU {
    v16bf h;
    unsigned int w[8];
};

#if __has_builtin(__builtin_amdgcn_global_load_async_to_lds_b128) && \
    __has_builtin(__builtin_amdgcn_s_wait_asynccnt)
#define HAVE_ASYNC 1
typedef int v4i_ __attribute__((vector_size(4 * sizeof(int))));
typedef __attribute__((address_space(1))) v4i_ GI4;
typedef __attribute__((address_space(3))) v4i_ LI4;
#else
#define HAVE_ASYNC 0
#endif

__device__ inline void cp16_async(const float* gp, float* lp) {
#if HAVE_ASYNC
    __builtin_amdgcn_global_load_async_to_lds_b128((GI4*)gp, (LI4*)lp, 0, 0);
#else
    *(float4*)lp = *(const float4*)gp;
#endif
}
__device__ inline void async_wait_all() {
#if HAVE_ASYNC
    __builtin_amdgcn_s_wait_asynccnt(0);
#endif
}

// float -> bf16 by truncation (cheap; 1 VALU op)
__device__ inline unsigned short f2bf_t(float x) {
    return (unsigned short)(__float_as_uint(x) >> 16);
}
// pack two floats' bf16 truncations into one dword: single v_perm_b32
__device__ inline unsigned int pk2t(float a, float b) {
    return __builtin_amdgcn_perm(__float_as_uint(b), __float_as_uint(a), 0x07060302u);
}
__device__ inline v8f zero8() {
    v8f z;
#pragma unroll
    for (int j = 0; j < 8; j++) z[j] = 0.0f;
    return z;
}

__global__ __launch_bounds__(NTHREADS)
void gdr_chunked_wmma(const float* __restrict__ q, const float* __restrict__ k,
                      const float* __restrict__ v, const float* __restrict__ a,
                      const float* __restrict__ bg, const float* __restrict__ A_log,
                      const float* __restrict__ dt_bias, float* __restrict__ out) {
    // double-buffered raw f32 chunk staging (filled by async global->LDS)
    __shared__ __align__(16) float rawQ[2][T_][D_ + 4];
    __shared__ __align__(16) float rawK[2][T_][D_ + 4];
    __shared__ __align__(16) float rawV[2][T_][MCOLS + 4];
    // bf16 operand staging; pair-contiguous layouts for b32 LDS reads
    __shared__ __align__(16) unsigned short sQ[T_][D_ + 8];       // [t][d] q~ = G_t*beta_t*scale*q/|q|
    __shared__ __align__(16) unsigned short sK[T_][D_ + 8];       // [s][d] k~ = k/(|k| G_s)
    __shared__ __align__(16) unsigned short sKT[D_][T_ + 8];      // [d][s] k~ transposed (update A-op)
    __shared__ __align__(16) unsigned short sVT[MCOLS][T_ + 8];   // [m][s] V transposed (B-op)
    __shared__ __align__(16) unsigned short sA[T_][T_ + 8];       // [t][s] masked scores
    __shared__ __align__(16) unsigned short sST[NWAVES][16][D_ + 8]; // [n][d] staged state (bf16)
    __shared__ float sGl[T_], sGc[T_], sBeta[T_], sGT;

    const int bid = blockIdx.x;
    const int msec = bid % MSPLIT;
    const int h = (bid / MSPLIT) % H_;
    const int b = bid / (MSPLIT * H_);
    const int mbase = msec * MCOLS;

    const int tid = threadIdx.x;
    const int wave = tid >> 5;
    const int lane = tid & 31;
    const int half = lane >> 4;
    const int ln = lane & 15;
    const int Klo = half << 3;          // K sub-offset from lane half
    const int wm = mbase + (wave << 4); // wave's first global m column

    // init: zero state staging (wave-local) and the statically-zero score tile (0,1)
    for (int i = lane; i < 16 * (D_ + 8); i += 32)
        ((unsigned short*)sST[wave])[i] = 0;
    for (int i = tid; i < 16 * 16; i += NTHREADS) sA[i >> 4][16 + (i & 15)] = 0;

    const float negA = -expf(A_log[h]);
    const float dtb = dt_bias[h];
    const float scale = rsqrtf((float)D_);

    const int row = tid >> 1;        // 0..31 chunk row handled by this thread
    const int cQ = (tid & 1) * 64;   // q/k column start (64 cols per thread)
    const int cV = (tid & 1) * 16;   // v column start within block slice

    // persistent state tiles: S[16r..16r+15 , wm..wm+15] as f32 C/D accumulators
    v8f St[8];
#pragma unroll
    for (int r = 0; r < 8; r++) St[r] = zero8();

    // issue async loads for chunk 0 into buffer 0
    {
        const size_t base0 = ((size_t)((b * S_ + row) * H_ + h)) * (size_t)D_;
#pragma unroll
        for (int i = 0; i < 16; i++) cp16_async(q + base0 + cQ + 4 * i, &rawQ[0][row][cQ + 4 * i]);
#pragma unroll
        for (int i = 0; i < 16; i++) cp16_async(k + base0 + cQ + 4 * i, &rawK[0][row][cQ + 4 * i]);
#pragma unroll
        for (int i = 0; i < 4; i++) cp16_async(v + base0 + mbase + cV + 4 * i, &rawV[0][row][cV + 4 * i]);
    }

    for (int ch = 0; ch < NC_; ch++) {
        const int t0 = ch * T_;
        const int p = ch & 1;

        // gates per chunk row (independent of raw chunk data)
        if (tid < T_) {
            const size_t gi = (size_t)((b * S_ + t0 + tid) * H_) + h;
            float av = a[gi] + dtb;
            float sp = (av > 20.f) ? av : log1pf(expf(av));
            sGl[tid] = negA * sp;
            sBeta[tid] = 1.f / (1.f + expf(-bg[gi]));
        }
        async_wait_all();   // my wave's async loads for chunk ch complete
        __syncthreads();    // A: raw[p] + sGl/sBeta visible block-wide

        // prefetch chunk ch+1 into the other buffer while we compute
        if (ch + 1 < NC_) {
            const size_t base1 = ((size_t)((b * S_ + t0 + T_ + row) * H_ + h)) * (size_t)D_;
#pragma unroll
            for (int i = 0; i < 16; i++) cp16_async(q + base1 + cQ + 4 * i, &rawQ[p ^ 1][row][cQ + 4 * i]);
#pragma unroll
            for (int i = 0; i < 16; i++) cp16_async(k + base1 + cQ + 4 * i, &rawK[p ^ 1][row][cQ + 4 * i]);
#pragma unroll
            for (int i = 0; i < 4; i++) cp16_async(v + base1 + mbase + cV + 4 * i, &rawV[p ^ 1][row][cV + 4 * i]);
        }

        // ---- pass 1: row sum-of-squares for q,k; stage V transposed ----
        float ssq = 0.f, ssk = 0.f;
        const float4* rq4 = (const float4*)(rawQ[p][row] + cQ);
        const float4* rk4 = (const float4*)(rawK[p][row] + cQ);
#pragma unroll
        for (int i = 0; i < 16; i++) {
            float4 t4 = rq4[i];
            ssq += t4.x * t4.x + t4.y * t4.y + t4.z * t4.z + t4.w * t4.w;
        }
#pragma unroll
        for (int i = 0; i < 16; i++) {
            float4 t4 = rk4[i];
            ssk += t4.x * t4.x + t4.y * t4.y + t4.z * t4.z + t4.w * t4.w;
        }
        {
            const float4* rv4 = (const float4*)(rawV[p][row] + cV);
#pragma unroll
            for (int i = 0; i < 4; i++) {
                float4 t4 = rv4[i];
                int c = cV + i * 4;
                sVT[c + 0][row] = f2bf_t(t4.x);
                sVT[c + 1][row] = f2bf_t(t4.y);
                sVT[c + 2][row] = f2bf_t(t4.z);
                sVT[c + 3][row] = f2bf_t(t4.w);
            }
        }
        ssq += __shfl_xor(ssq, 1);
        ssk += __shfl_xor(ssk, 1);
        const float invq = 1.f / fmaxf(sqrtf(ssq), 1e-12f);
        const float invk = 1.f / fmaxf(sqrtf(ssk), 1e-12f);

        if (tid < T_) {  // inclusive cumulative gate product within chunk
            float p2 = 1.f;
            for (int s2 = 0; s2 <= tid; s2++) p2 *= sGl[s2];
            sGc[tid] = p2;
            if (tid == T_ - 1) sGT = p2;
        }
        __syncthreads();  // B: sGc/sGT visible

        // ---- pass 2: stage scaled q~, k~ (row-major) + k~ transposed ----
        {
            const float Gt = sGc[row];
            const float sq_ = Gt * sBeta[row] * scale * invq;
            const float sk_ = invk / Gt;
#pragma unroll
            for (int i = 0; i < 16; i++) {
                float4 t4 = rq4[i];
                float4 u4 = rk4[i];
                int c = cQ + i * 4;
                uint2 qu, ku;
                qu.x = pk2t(t4.x * sq_, t4.y * sq_);
                qu.y = pk2t(t4.z * sq_, t4.w * sq_);
                ku.x = pk2t(u4.x * sk_, u4.y * sk_);
                ku.y = pk2t(u4.z * sk_, u4.w * sk_);
                *(uint2*)&sQ[row][c] = qu;   // ds_store_b64
                *(uint2*)&sK[row][c] = ku;   // ds_store_b64
                sKT[c + 0][row] = (unsigned short)ku.x;
                sKT[c + 1][row] = (unsigned short)(ku.x >> 16);
                sKT[c + 2][row] = (unsigned short)ku.y;
                sKT[c + 3][row] = (unsigned short)(ku.y >> 16);
            }
        }
        __syncthreads();  // C: all bf16 staging visible

        // ---- scores: 32x32 = 4 tiles, split across waves; tile(0,1) is static 0 ----
        // wave0 computes unmasked tile (mi=1,si=0); wave1 computes diagonal tiles (0,0),(1,1)
        if (wave == 0) {
            v8f t10 = zero8();
#pragma unroll
            for (int c2 = 0; c2 < 4; c2++) {
                HU qa, kb;
#pragma unroll
                for (int vv = 0; vv < 8; vv++) {
                    int K0 = c2 * 32 + (vv >> 2) * 16 + Klo + (vv & 3) * 2;
                    qa.w[vv] = *(const unsigned int*)&sQ[16 + ln][K0];
                    kb.w[vv] = *(const unsigned int*)&sK[ln][K0];
                }
                t10 = __builtin_amdgcn_wmma_f32_16x16x32_bf16(false, qa.h, false, kb.h,
                                                              (short)0, t10, false, false);
            }
#pragma unroll
            for (int j = 0; j < 8; j++) {
                int M = j + (half << 3);
                sA[16 + M][ln] = f2bf_t(t10[j]);   // s<16<=t: never masked
            }
        } else {
#pragma unroll
            for (int dtile = 0; dtile < 2; dtile++) {  // tiles (0,0) and (1,1)
                const int o16 = dtile << 4;
                v8f td = zero8();
#pragma unroll
                for (int c2 = 0; c2 < 4; c2++) {
                    HU qa, kb;
#pragma unroll
                    for (int vv = 0; vv < 8; vv++) {
                        int K0 = c2 * 32 + (vv >> 2) * 16 + Klo + (vv & 3) * 2;
                        qa.w[vv] = *(const unsigned int*)&sQ[o16 + ln][K0];
                        kb.w[vv] = *(const unsigned int*)&sK[o16 + ln][K0];
                    }
                    td = __builtin_amdgcn_wmma_f32_16x16x32_bf16(false, qa.h, false, kb.h,
                                                                 (short)0, td, false, false);
                }
#pragma unroll
                for (int j = 0; j < 8; j++) {
                    int M = j + (half << 3);
                    sA[o16 + M][o16 + ln] = f2bf_t((ln <= M) ? td[j] : 0.0f);
                }
            }
        }
        __syncthreads();  // D: masked scores visible

        // ---- V tile B-operand (full K=32, contiguous b32 reads; reused 10x) ----
        HU vB;
#pragma unroll
        for (int vv = 0; vv < 8; vv++) {
            int K0 = (vv >> 2) * 16 + Klo + (vv & 3) * 2;
            vB.w[vv] = *(const unsigned int*)&sVT[(wave << 4) + ln][K0];
        }

        // ---- output: O[mi] = A_masked[mi,:] @ V + Q~[mi] @ S_0 ----
        v8f o0 = zero8(), o1 = zero8();
        {
            HU a0, a1;
#pragma unroll
            for (int vv = 0; vv < 8; vv++) {
                int K0 = (vv >> 2) * 16 + Klo + (vv & 3) * 2;
                a0.w[vv] = *(const unsigned int*)&sA[ln][K0];
                a1.w[vv] = *(const unsigned int*)&sA[16 + ln][K0];
            }
            o0 = __builtin_amdgcn_wmma_f32_16x16x32_bf16(false, a0.h, false, vB.h,
                                                         (short)0, o0, false, false);
            o1 = __builtin_amdgcn_wmma_f32_16x16x32_bf16(false, a1.h, false, vB.h,
                                                         (short)0, o1, false, false);
        }
#pragma unroll
        for (int c2 = 0; c2 < 4; c2++) {
            HU sB, q0, q1;
#pragma unroll
            for (int vv = 0; vv < 8; vv++) {
                int K0 = c2 * 32 + (vv >> 2) * 16 + Klo + (vv & 3) * 2;
                sB.w[vv] = *(const unsigned int*)&sST[wave][ln][K0];
                q0.w[vv] = *(const unsigned int*)&sQ[ln][K0];
                q1.w[vv] = *(const unsigned int*)&sQ[16 + ln][K0];
            }
            o0 = __builtin_amdgcn_wmma_f32_16x16x32_bf16(false, q0.h, false, sB.h,
                                                         (short)0, o0, false, false);
            o1 = __builtin_amdgcn_wmma_f32_16x16x32_bf16(false, q1.h, false, sB.h,
                                                         (short)0, o1, false, false);
        }

        // ---- store O tiles ----
        {
            const size_t ob = ((size_t)((b * S_ + t0) * H_ + h)) * (size_t)D_ + wm + ln;
#pragma unroll
            for (int j = 0; j < 8; j++) {
                int t = j + (half << 3);
                out[ob + (size_t)t * (H_ * D_)] = o0[j];
                out[ob + (size_t)(16 + t) * (H_ * D_)] = o1[j];
            }
        }

        // ---- state update: S = G_T * (S + K~^T V), full K=32 ----
        const float gT = sGT;
#pragma unroll
        for (int r = 0; r < 8; r++) {
            HU kA;
#pragma unroll
            for (int vv = 0; vv < 8; vv++) {
                int K0 = (vv >> 2) * 16 + Klo + (vv & 3) * 2;
                kA.w[vv] = *(const unsigned int*)&sKT[(r << 4) + ln][K0];
            }
            v8f d2 = __builtin_amdgcn_wmma_f32_16x16x32_bf16(false, kA.h, false, vB.h,
                                                             (short)0, St[r], false, false);
#pragma unroll
            for (int j = 0; j < 8; j++) St[r][j] = d2[j] * gT;
        }

        // ---- stage updated state slice (wave-local, packed b128 stores) ----
#pragma unroll
        for (int r = 0; r < 8; r++) {
            const int d0 = (r << 4) + (half << 3);
            uint4 u;
            u.x = pk2t(St[r][0], St[r][1]);
            u.y = pk2t(St[r][2], St[r][3]);
            u.z = pk2t(St[r][4], St[r][5]);
            u.w = pk2t(St[r][6], St[r][7]);
            *(uint4*)&sST[wave][ln][d0] = u;   // ds_store_b128
        }
        // sST[wave] is written and read only by this wave; loop barriers order
        // the cross-wave arrays.
    }
}

extern "C" void kernel_launch(void* const* d_in, const int* in_sizes, int n_in,
                              void* d_out, int out_size, void* d_ws, size_t ws_size,
                              hipStream_t stream) {
    const float* q = (const float*)d_in[0];
    const float* k = (const float*)d_in[1];
    const float* v = (const float*)d_in[2];
    const float* a = (const float*)d_in[3];
    const float* bg = (const float*)d_in[4];
    const float* A_log = (const float*)d_in[5];
    const float* dt_bias = (const float*)d_in[6];
    float* out = (float*)d_out;

    dim3 grid(B_ * H_ * MSPLIT);
    dim3 block(NTHREADS);
    hipLaunchKernelGGL(gdr_chunked_wmma, grid, block, 0, stream,
                       q, k, v, a, bg, A_log, dt_bias, out);
}
